// PoolGraph_16621523436297
// MI455X (gfx1250) — compile-verified, matrix operand order checked by reading, not run on hardware
//
#include <hip/hip_runtime.h>
#include <hip/hip_bf16.h>
#include <math.h>

// ---------------------------------------------------------------------------
// PoolGraph pipeline for MI455X (gfx1250, wave32).
//   N=4096 nodes, F=64 feats, E=131072 edges, K=2048 kept nodes, EPS=0.1
// CSR adjacency (no dense 16MB matrix), single-workgroup bitonic top-k in
// LDS, per-node fractal-dim with LDS hash buffer + pow2-sized bitonic unique
// count, and the x@W GEMV on the f32 WMMA path (v_wmma_f32_16x16x4_f32,
// exact f32 accumulation matching the reference).
// ---------------------------------------------------------------------------

#define NN 4096
#define FF 64
#define EE 131072
#define KK 2048
#define INV_EPS 10.0f
#define FD_MAXD 512          // safe cap: max degree ~ Binom(131072,1/4096) << 512

typedef __attribute__((ext_vector_type(16))) _Float16 v16h;
typedef __attribute__((ext_vector_type(8)))  float    v8f;
typedef __attribute__((ext_vector_type(2)))  float    v2f;

// deterministic stand-in for the host-side random 62-bit hash multipliers
__device__ __forceinline__ long long hash_mult(int f) {
  unsigned long long z = (unsigned long long)(f + 1) * 0x9E3779B97F4A7C15ULL;
  z ^= z >> 30; z *= 0xBF58476D1CE4E5B9ULL;
  z ^= z >> 27; z *= 0x94D049BB133111EBULL;
  z ^= z >> 31;
  return (long long)((z & 0x3FFFFFFFFFFFFFFFULL) | 1ULL);
}

// --- degree (dst) + CSR row counts (src) -----------------------------------
__global__ void k_count_deg(const long long* __restrict__ ei,
                            unsigned* __restrict__ deg,
                            unsigned* __restrict__ cnt1) {
  int e = blockIdx.x * blockDim.x + threadIdx.x;
  if (e >= EE) return;
  int s = (int)ei[e];
  int d = (int)ei[EE + e];
  atomicAdd(&cnt1[s], 1u);
  atomicAdd(&deg[d], 1u);
}

// --- exclusive scan of up to 4096 counts in one 1024-thread workgroup ------
__global__ __launch_bounds__(1024)
void k_scan(const unsigned* __restrict__ cnt, unsigned* __restrict__ off,
            unsigned* __restrict__ cur, int n) {
  __shared__ unsigned s[1024];
  int tid = threadIdx.x;
  int items = (n + 1023) >> 10;       // 4 for n=4096, 2 for n=2048
  unsigned loc[4];
  unsigned run = 0;
#pragma unroll
  for (int it = 0; it < 4; ++it) {
    if (it < items) {
      int idx = tid * items + it;
      unsigned v = (idx < n) ? cnt[idx] : 0u;
      loc[it] = run; run += v;
    }
  }
  s[tid] = run; __syncthreads();
  for (int o = 1; o < 1024; o <<= 1) {
    unsigned v = (tid >= o) ? s[tid - o] : 0u;
    __syncthreads();
    s[tid] += v;
    __syncthreads();
  }
  unsigned base = s[tid] - run;       // exclusive prefix
#pragma unroll
  for (int it = 0; it < 4; ++it) {
    if (it < items) {
      int idx = tid * items + it;
      if (idx < n) { unsigned v = base + loc[it]; off[idx] = v; cur[idx] = v; }
    }
  }
  if (tid == 1023) off[n] = base + run;
}

// --- CSR fill (before graph) -----------------------------------------------
__global__ void k_fill1(const long long* __restrict__ ei,
                        unsigned* __restrict__ cur1,
                        unsigned* __restrict__ col1) {
  int e = blockIdx.x * blockDim.x + threadIdx.x;
  if (e >= EE) return;
  int s = (int)ei[e];
  int d = (int)ei[EE + e];
  unsigned p = atomicAdd(&cur1[s], 1u);
  col1[p] = (unsigned)d;
}

// --- h = x @ W : one wave computes 16 rows ---------------------------------
// Preferred path: V_WMMA_F32_16X16X4_F32 (pure-f32 WMMA, exact accumulate).
// Fallback path (if builtin undeclared): confirmed f32_16x16x32_f16.
__global__ __launch_bounds__(32)
void k_gemv_wmma(const float* __restrict__ x, const float* __restrict__ W,
                 float* __restrict__ h) {
  int lane = threadIdx.x;                    // 0..31, wave32
  int tile = blockIdx.x;                     // 256 tiles of 16 rows
  int row  = tile * 16 + (lane & 15);
  bool hi   = lane >= 16;
  bool col0 = (lane & 15) == 0;
  const float* xr = x + (size_t)row * FF;
  v8f c = {};

#if __has_builtin(__builtin_amdgcn_wmma_f32_16x16x4_f32)
  // ISA 32-bit A 16x4 layout: VGPR0 = K(0|2), VGPR1 = K(1|3) for lanes lo|hi.
  // B 4x16: only column 0 live (GEMV): VGPR0 = K(0|1), VGPR1 = K(2|3) for lanes lo|hi (differs from A).
#pragma unroll
  for (int k = 0; k < FF; k += 4) {
    v2f a, b;
    a[0] = xr[k + (hi ? 2 : 0)];
    a[1] = xr[k + (hi ? 3 : 1)];
    float w0 = W[k + (hi ? 1 : 0)];          // unconditional load: no exec branch
    float w1 = W[k + (hi ? 3 : 2)];
    b[0] = col0 ? w0 : 0.0f;
    b[1] = col0 ? w1 : 0.0f;
    c = __builtin_amdgcn_wmma_f32_16x16x4_f32(false, a, false, b,
                                              (short)0, c, false, false);
  }
#else
  // f16 inputs / f32 accumulate fallback (codegen-confirmed builtin)
  int ka = hi ? 8 : 0;
  v16h a0, a1;
#pragma unroll
  for (int j = 0; j < 8; ++j) {
    a0[j]     = (_Float16)xr[ka + j];
    a0[8 + j] = (_Float16)xr[16 + ka + j];
    a1[j]     = (_Float16)xr[32 + ka + j];
    a1[8 + j] = (_Float16)xr[48 + ka + j];
  }
  int kb = hi ? 16 : 0;
  v16h b0, b1;
#pragma unroll
  for (int j = 0; j < 16; ++j) {
    float w0 = W[kb + j], w1 = W[32 + kb + j];
    b0[j] = col0 ? (_Float16)w0 : (_Float16)0.0f;
    b1[j] = col0 ? (_Float16)w1 : (_Float16)0.0f;
  }
  c = __builtin_amdgcn_wmma_f32_16x16x32_f16(false, a0, false, b0,
                                             (short)0, c, false, false);
  c = __builtin_amdgcn_wmma_f32_16x16x32_f16(false, a1, false, b1,
                                             (short)0, c, false, false);
#endif

  if (col0) {
    int rbase = tile * 16 + (hi ? 8 : 0);    // C: column N=0 lives in lanes 0/16
#pragma unroll
    for (int r = 0; r < 8; ++r) h[rbase + r] = c[r];
  }
}

// --- dinv = (deg+selfloop)^-1/2 --------------------------------------------
__global__ void k_dinv(const unsigned* __restrict__ deg, float* __restrict__ dinv) {
  int i = blockIdx.x * blockDim.x + threadIdx.x;
  if (i < NN) dinv[i] = rsqrtf((float)(deg[i] + 1u));
}

// --- edge scatter: agg[dst] += dinv[s]*dinv[d]*h[s] ------------------------
__global__ void k_scatter(const long long* __restrict__ ei,
                          const float* __restrict__ dinv,
                          const float* __restrict__ h,
                          float* __restrict__ agg) {
  int e = blockIdx.x * blockDim.x + threadIdx.x;
  if (e >= EE) return;
  int s = (int)ei[e];
  int d = (int)ei[EE + e];
  atomicAdd(&agg[d], dinv[s] * dinv[d] * h[s]);
}

// --- score = tanh(agg + selfloop + b) --------------------------------------
__global__ void k_score(const float* __restrict__ agg, const float* __restrict__ dinv,
                        const float* __restrict__ h, const float* __restrict__ b,
                        float* __restrict__ score) {
  int i = blockIdx.x * blockDim.x + threadIdx.x;
  if (i < NN) score[i] = tanhf(agg[i] + dinv[i] * dinv[i] * h[i] + b[0]);
}

// --- top-k: full 4096 bitonic sort (desc score, asc idx) in 32KB LDS -------
__global__ __launch_bounds__(1024)
void k_topk(const float* __restrict__ score, int* __restrict__ perm,
            int* __restrict__ remap) {
  __shared__ unsigned long long key[NN];     // 32 KB of the 320 KB WGP LDS
  int tid = threadIdx.x;
  for (int i = tid; i < NN; i += 1024) {
    unsigned u = __float_as_uint(score[i]);
    unsigned m = (u & 0x80000000u) ? ~u : (u | 0x80000000u);   // asc-monotone
    key[i] = ((unsigned long long)(m ^ 0xFFFFFFFFu) << 32) | (unsigned)i;
  }
  __syncthreads();
  for (int k = 2; k <= NN; k <<= 1)
    for (int j = k >> 1; j > 0; j >>= 1) {
      for (int idx = tid; idx < NN; idx += 1024) {
        int ixj = idx ^ j;
        if (ixj > idx) {
          unsigned long long a = key[idx], b2 = key[ixj];
          bool up = ((idx & k) == 0);
          if ((a > b2) == up) { key[idx] = b2; key[ixj] = a; }
        }
      }
      __syncthreads();
    }
  for (int t = tid; t < NN; t += 1024) {
    int node = (int)(key[t] & 0xFFFFFFFFu);
    if (t < KK) perm[t] = node;
    remap[node] = (t < KK) ? t : -1;
  }
}

// --- gather + gate: x_p = x[perm]*score[perm]; batch_p = 0 -----------------
__global__ __launch_bounds__(64)
void k_gather(const float* __restrict__ x, const float* __restrict__ score,
              const int* __restrict__ perm, float* __restrict__ xp,
              float* __restrict__ batp) {
  int t = blockIdx.x, f = threadIdx.x;
  int r = perm[t];
  float s = score[r];
  xp[(size_t)t * FF + f] = x[(size_t)r * FF + f] * s;
  if (f == 0) batp[t] = 0.0f;
}

// --- filter_adj: remap edges, emit new_ei (float), count CSR2 rows ---------
__global__ void k_filter(const long long* __restrict__ ei,
                         const int* __restrict__ remap,
                         float* __restrict__ out_ei,
                         unsigned* __restrict__ cnt2) {
  int e = blockIdx.x * blockDim.x + threadIdx.x;
  if (e >= EE) return;
  int s = (int)ei[e];
  int d = (int)ei[EE + e];
  int rs = remap[s], rd = remap[d];
  bool valid = (rs >= 0) && (rd >= 0);
  out_ei[e]      = valid ? (float)rs : -1.0f;
  out_ei[EE + e] = valid ? (float)rd : -1.0f;
  if (valid) atomicAdd(&cnt2[rs], 1u);
}

__global__ void k_fill2(const long long* __restrict__ ei,
                        const int* __restrict__ remap,
                        unsigned* __restrict__ cur2,
                        unsigned* __restrict__ col2) {
  int e = blockIdx.x * blockDim.x + threadIdx.x;
  if (e >= EE) return;
  int rs = remap[(int)ei[e]];
  int rd = remap[(int)ei[EE + e]];
  if (rs >= 0 && rd >= 0) {
    unsigned p = atomicAdd(&cur2[rs], 1u);
    col2[p] = (unsigned)rd;
  }
}

// --- fractal dim: one 128-thread block per node, all work in LDS -----------
__global__ __launch_bounds__(128)
void k_fractal(const float* __restrict__ coords, const unsigned* __restrict__ off,
               const unsigned* __restrict__ col, float* __restrict__ out) {
  __shared__ float     s_part[2][FF];
  __shared__ float     s_min[FF];
  __shared__ long long s_hash[FD_MAXD];      // 4 KB
  __shared__ int       s_cnt;
  const long long SENT = 0x7FFFFFFFFFFFFFFFLL;

  int i = blockIdx.x, tid = threadIdx.x;
  unsigned beg = off[i];
  int d = (int)(off[i + 1] - beg);
  if (d > FD_MAXD) d = FD_MAXD;
  if (d == 0) { if (tid == 0) out[i] = 0.0f; return; }
  int n2 = 2;                                // pow2 pad (mean degree 32 -> 32/64)
  while (n2 < d) n2 <<= 1;

  // phase A: per-feature min over neighbors (duplicates harmless)
  int f = tid & 63, p = tid >> 6;
  float m = INFINITY;
  for (int j = p; j < d; j += 2) {
    if (j + 2 < d)   // prefetch next neighbor row (global_prefetch_b8)
      __builtin_prefetch(coords + (size_t)col[beg + j + 2] * FF, 0, 1);
    m = fminf(m, coords[(size_t)col[beg + j] * FF + f]);
  }
  s_part[p][f] = m;
  __syncthreads();
  if (tid < FF) s_min[tid] = fminf(s_part[0][tid], s_part[1][tid]);
  if (tid == 0) s_cnt = 0;
  __syncthreads();

  // phase B: 62-bit box hash per neighbor (duplicate edges -> equal hashes)
  for (int j = tid; j < n2; j += 128) {
    long long hv = SENT;
    if (j < d) {
      const float* c = coords + (size_t)col[beg + j] * FF;
      long long acc = 0;
#pragma unroll 8
      for (int ff = 0; ff < FF; ++ff) {
        long long box = (long long)floorf((c[ff] - s_min[ff]) * INV_EPS);
        acc += box * hash_mult(ff);
      }
      hv = acc;
    }
    s_hash[j] = hv;
  }
  __syncthreads();

  // phase C: bitonic sort n2 keys in LDS, then unique count
  for (int k = 2; k <= n2; k <<= 1)
    for (int j = k >> 1; j > 0; j >>= 1) {
      for (int idx = tid; idx < n2; idx += 128) {
        int ixj = idx ^ j;
        if (ixj > idx) {
          long long a = s_hash[idx], b2 = s_hash[ixj];
          bool up = ((idx & k) == 0);
          if ((a > b2) == up) { s_hash[idx] = b2; s_hash[ixj] = a; }
        }
      }
      __syncthreads();
    }
  int local = 0;
  for (int idx = tid; idx < n2; idx += 128) {
    long long v = s_hash[idx];
    if (v != SENT && (idx == 0 || s_hash[idx - 1] != v)) ++local;
  }
  atomicAdd(&s_cnt, local);
  __syncthreads();
  if (tid == 0) out[i] = (float)s_cnt;
}

// ---------------------------------------------------------------------------
extern "C" void kernel_launch(void* const* d_in, const int* in_sizes, int n_in,
                              void* d_out, int out_size, void* d_ws, size_t ws_size,
                              hipStream_t stream) {
  const float*      x    = (const float*)d_in[0];
  const long long*  ei   = (const long long*)d_in[1];   // int64 edge_index [2,E]
  /* batch (d_in[2]) is all zeros -> unused */
  const float*      W    = (const float*)d_in[3];
  const float*      bptr = (const float*)d_in[4];

  float* out = (float*)d_out;
  // output layout (flat, return order): x_p | new_ei | batch_p | frac_b | frac_a
  const size_t o_xp = 0;
  const size_t o_ei = (size_t)KK * FF;                  // 131072
  const size_t o_bt = o_ei + 2 * (size_t)EE;            // 393216
  const size_t o_fb = o_bt + KK;                        // 395264
  const size_t o_fa = o_fb + NN;                        // 399360

  // workspace carve-up (all 4-byte aligned)
  char* w = (char*)d_ws;
  float*    h_    = (float*)w;            w += (size_t)NN * 4;
  float*    dinv_ = (float*)w;            w += (size_t)NN * 4;
  float*    agg_  = (float*)w;            w += (size_t)NN * 4;
  float*    sc_   = (float*)w;            w += (size_t)NN * 4;
  unsigned* deg_  = (unsigned*)w;         w += (size_t)NN * 4;
  unsigned* cnt1  = (unsigned*)w;         w += (size_t)NN * 4;
  unsigned* off1  = (unsigned*)w;         w += (size_t)(NN + 1) * 4;
  unsigned* cur1  = (unsigned*)w;         w += (size_t)NN * 4;
  unsigned* col1  = (unsigned*)w;         w += (size_t)EE * 4;
  int*      perm  = (int*)w;              w += (size_t)KK * 4;
  int*      remap = (int*)w;              w += (size_t)NN * 4;
  unsigned* cnt2  = (unsigned*)w;         w += (size_t)KK * 4;
  unsigned* off2  = (unsigned*)w;         w += (size_t)(KK + 1) * 4;
  unsigned* cur2  = (unsigned*)w;         w += (size_t)KK * 4;
  unsigned* col2  = (unsigned*)w;         /* += EE*4 */

  hipMemsetAsync(deg_, 0, (size_t)NN * 4, stream);
  hipMemsetAsync(cnt1, 0, (size_t)NN * 4, stream);
  hipMemsetAsync(agg_, 0, (size_t)NN * 4, stream);
  hipMemsetAsync(cnt2, 0, (size_t)KK * 4, stream);

  k_count_deg<<<EE / 256, 256, 0, stream>>>(ei, deg_, cnt1);
  k_scan<<<1, 1024, 0, stream>>>(cnt1, off1, cur1, NN);
  k_fill1<<<EE / 256, 256, 0, stream>>>(ei, cur1, col1);

  k_gemv_wmma<<<NN / 16, 32, 0, stream>>>(x, W, h_);
  k_dinv<<<NN / 256, 256, 0, stream>>>(deg_, dinv_);
  k_scatter<<<EE / 256, 256, 0, stream>>>(ei, dinv_, h_, agg_);
  k_score<<<NN / 256, 256, 0, stream>>>(agg_, dinv_, h_, bptr, sc_);

  k_topk<<<1, 1024, 0, stream>>>(sc_, perm, remap);
  k_gather<<<KK, 64, 0, stream>>>(x, sc_, perm, out + o_xp, out + o_bt);

  k_filter<<<EE / 256, 256, 0, stream>>>(ei, remap, out + o_ei, cnt2);
  k_scan<<<1, 1024, 0, stream>>>(cnt2, off2, cur2, KK);
  k_fill2<<<EE / 256, 256, 0, stream>>>(ei, remap, cur2, col2);

  k_fractal<<<NN, 128, 0, stream>>>(x, off1, col1, out + o_fb);
  k_fractal<<<KK, 128, 0, stream>>>(out + o_xp, off2, col2, out + o_fa);
}
